// NRIMD_25460566130846
// MI455X (gfx1250) — compile-verified, matrix operand order checked by reading, not run on hardware
//
#include <hip/hip_runtime.h>
#include <hip/hip_bf16.h>

typedef __attribute__((ext_vector_type(16))) _Float16 v16h;
typedef __attribute__((ext_vector_type(8)))  _Float16 v8h;
typedef __attribute__((ext_vector_type(8)))  float    v8f;

#define DEV static __device__ __forceinline__

// ---------------- problem constants ----------------
constexpr int  B_  = 8;
constexpr int  T_  = 50;
constexpr int  N_  = 256;
constexpr int  D_  = 12;
constexpr int  H_  = 6;
constexpr int  ET_ = 2;
constexpr int  E_  = N_ * (N_ - 1);          // 65280
constexpr long ROWS = (long)B_ * E_;         // 522240 edge rows
constexpr int  NTILES = (int)(ROWS / 16);    // 32640 (exact)
constexpr int  PRED_SZ = B_ * (T_ - 1) * N_ * D_;  // 1204224
constexpr int  EDGE_SZ = B_ * E_ * ET_;            // 1044480
constexpr int  XE_K = 608;                   // 600 + bias row + pad (19 chunks of 32)

// WMMA 16x16x32 f16 fragment k-map: lane L (hi=L>=16, ln=L&15),
//   A: row m=ln, half i -> k = i + 8*(i>=8) + 8*hi   (each half-wave holds two 8-long K runs)
//   B: col n=ln, same k map.  C f32: c[j] = C[j+8*hi][ln]
DEV int wk_map(int i, int hi) { return i + ((i >= 8) ? 8 : 0) + (hi ? 8 : 0); }

DEV void edge_rs(int e, int& recv, int& send) {
  recv = e / (N_ - 1);
  int k = e - recv * (N_ - 1);
  send = (k < recv) ? k : k + 1;
}

DEV float sigmoidf_(float x) { return 1.f / (1.f + __expf(-x)); }

// CDNA5 hardware transcendental tanh (v_tanh_f32); TRANS ops co-execute with VALU
DEV float fast_tanh(float x) {
#if __has_builtin(__builtin_amdgcn_tanhf)
  return __builtin_amdgcn_tanhf(x);
#else
  float y;
  asm("v_tanh_f32 %0, %1" : "=v"(y) : "v"(x));
  return y;
#endif
}

// ---------------- prep: pack all weight (B-operand) fragments once ----------------
__global__ void k_prep(const float* __restrict__ m1W1, const float* __restrict__ m1b1,
                       const float* __restrict__ m2W1, const float* __restrict__ m2b1,
                       const float* __restrict__ m2W2, const float* __restrict__ m2b2,
                       const float* __restrict__ m4W1, const float* __restrict__ m4b1,
                       const float* __restrict__ m4W2, const float* __restrict__ m4b2,
                       const float* __restrict__ W1m,  const float* __restrict__ b1m,
                       const float* __restrict__ W2m,  const float* __restrict__ b2m,
                       _Float16* __restrict__ FB1, _Float16* __restrict__ F2a,
                       _Float16* __restrict__ F2b, _Float16* __restrict__ F4a,
                       _Float16* __restrict__ F4b, _Float16* __restrict__ FMa,
                       _Float16* __restrict__ FMb) {
  const int tid = blockIdx.x * 256 + threadIdx.x;
  if (tid >= 19 * 32 + 6 * 32) return;
  int lane, sel, kc = 0;
  if (tid < 19 * 32) { kc = tid >> 5; lane = tid & 31; sel = 0; }
  else { int u = tid - 19 * 32; sel = 1 + (u >> 5); lane = u & 31; }
  const int hi = lane >> 4, ln = lane & 15;
  _Float16* dst;
  switch (sel) {
    case 0: dst = FB1 + (kc * 32 + lane) * 16; break;
    case 1: dst = F2a + lane * 16; break;
    case 2: dst = F2b + lane * 16; break;
    case 3: dst = F4a + lane * 16; break;
    case 4: dst = F4b + lane * 16; break;
    case 5: dst = FMa + lane * 16; break;
    default: dst = FMb + lane * 16; break;
  }
  for (int i = 0; i < 16; ++i) {
    const int k = wk_map(i, hi);
    float bv = 0.f;
    switch (sel) {
      case 0: { const int kg = kc * 32 + k;
                if (kg < 600 && ln < H_) bv = m1W1[kg * H_ + ln];
                else if (kg == 600 && ln < H_) bv = m1b1[ln]; } break;
      case 1: if (ln < H_) { if (k < 12) bv = m2W1[k * H_ + ln]; else if (k == 12) bv = m2b1[ln]; } break;
      case 2: if (ln < H_) { if (k < 6)  bv = m2W2[k * H_ + ln]; else if (k == 6)  bv = m2b2[ln]; } break;
      case 3: if (ln < H_) { if (k < 18) bv = m4W1[k * H_ + ln]; else if (k == 18) bv = m4b1[ln]; } break;
      case 4: if (ln < H_) { if (k < 6)  bv = m4W2[k * H_ + ln]; else if (k == 6)  bv = m4b2[ln]; } break;
      case 5: if (ln < 12) { const int et = ln / 6, nc = ln % 6;            // cols 0-5 type0, 6-11 type1
                if (k < 12) bv = W1m[et * 72 + k * H_ + nc];
                else if (k == 12) bv = b1m[et * H_ + nc]; } break;
      default: {                                                            // block-diagonal fc2
                if (k < 6 && ln < 6) bv = W2m[k * H_ + ln];
                else if (k >= 6 && k < 12 && ln >= 6 && ln < 12) bv = W2m[36 + (k - 6) * H_ + (ln - 6)];
                else if (k == 12 && ln < 12) bv = b2m[(ln / 6) * H_ + (ln % 6)]; } break;
    }
    dst[i] = (_Float16)bv;
  }
}

// zero hidden state (f32 + padded f16 mirror)
__global__ void k_init(float* __restrict__ hid, _Float16* __restrict__ hidh8) {
  const int i = blockIdx.x * 256 + threadIdx.x;
  if (i < B_ * N_ * H_) hid[i] = 0.f;
  if (i < B_ * N_ * 8)  hidh8[i] = (_Float16)0.f;
}

// x [B,T,N,D] f32 -> xe [B*N][608] f16 (k=600 -> 1.0 bias row, 601..607 -> 0)
__global__ void k_xpose(const float* __restrict__ x, _Float16* __restrict__ xe) {
  const int idx = blockIdx.x * 256 + threadIdx.x;
  if (idx >= B_ * N_ * XE_K) return;
  const int row = idx / XE_K, k = idx - row * XE_K;
  const int b = row >> 8, n = row & 255;
  float v = 0.f;
  if (k < 600) v = x[((b * T_ + k / D_) * N_ + n) * D_ + (k % D_)];
  else if (k == 600) v = 1.f;
  xe[idx] = (_Float16)v;
}

// encoder mlp1: 2048 rows x K=608 via WMMA, contiguous vector A loads
__global__ void k_mlp1(const _Float16* __restrict__ xe, const _Float16* __restrict__ FB1,
                       const float* __restrict__ W2, const float* __restrict__ b2,
                       _Float16* __restrict__ hN8) {
  __shared__ __align__(16) _Float16 lds[4][16][16];
  const int wave = threadIdx.x >> 5, lane = threadIdx.x & 31;
  const int hi = lane >> 4, ln = lane & 15;
  const int rg0 = (blockIdx.x * 4 + wave) * 16;
  const int rg = rg0 + ln;
  const v8h*  rowp = (const v8h*)xe + rg * (XE_K / 8);     // 76 v8h per row
  const v16h* fbp  = (const v16h*)FB1;
  v8f c = {};
  for (int kc = 0; kc < 19; ++kc) {
    const v8h alo = rowp[kc * 4 + hi];                     // k = kc*32 + hi*8 .. +8
    const v8h ahi = rowp[kc * 4 + 2 + hi];                 // k = kc*32 + 16 + hi*8 .. +8
    v16h a;
#pragma unroll
    for (int i = 0; i < 8; ++i) { a[i] = alo[i]; a[8 + i] = ahi[i]; }
    const v16h bf = fbp[kc * 32 + lane];
    c = __builtin_amdgcn_wmma_f32_16x16x32_f16(false, a, false, bf, (short)0, c, false, false);
  }
#pragma unroll
  for (int j = 0; j < 8; ++j) { float v = c[j]; lds[wave][j + 8 * hi][ln] = (_Float16)(v > 0.f ? v : 0.f); }
  __syncthreads();
  if (hi == 0) {                                            // tiny 6x6 second layer
    const int m = ln, ro = rg0 + m;
#pragma unroll
    for (int col = 0; col < H_; ++col) {
      float s = b2[col];
#pragma unroll
      for (int k = 0; k < H_; ++k) s += (float)lds[wave][m][k] * W2[k * H_ + col];
      hN8[ro * 8 + col] = (_Float16)s;
    }
    hN8[ro * 8 + 6] = (_Float16)0.f; hN8[ro * 8 + 7] = (_Float16)0.f;
  }
}

// encoder mlp2 over edges: A = [h[recv] | h[send] | 1], two chained WMMAs
__global__ void k_mlp2(const _Float16* __restrict__ hN8, const _Float16* __restrict__ F2a,
                       const _Float16* __restrict__ F2b, _Float16* __restrict__ eN8) {
  __shared__ __align__(16) _Float16 lds[4][16][16];
  const int wave = threadIdx.x >> 5, lane = threadIdx.x & 31;
  const int hi = lane >> 4, ln = lane & 15;
  const long rg0 = ((long)blockIdx.x * 4 + wave) * 16;
  const long rg = rg0 + ln;
  const int b = (int)(rg / E_), e = (int)(rg % E_);
  int recv, send; edge_rs(e, recv, send);
  const v8h* hp = (const v8h*)hN8;
  const v8h hr = hp[b * N_ + recv];
  const v8h hs = hp[b * N_ + send];
  v16h a = {};
  if (hi == 0) {                                           // k0..7 = recv[0..5], send[0..1]
#pragma unroll
    for (int i = 0; i < 6; ++i) a[i] = hr[i];
    a[6] = hs[0]; a[7] = hs[1];
  } else {                                                 // k8..15 = send[2..5], 1, 0...
#pragma unroll
    for (int i = 0; i < 4; ++i) a[i] = hs[2 + i];
    a[4] = (_Float16)1.f;
  }
  v8f c = {};
  c = __builtin_amdgcn_wmma_f32_16x16x32_f16(false, a, false, ((const v16h*)F2a)[lane], (short)0, c, false, false);
#pragma unroll
  for (int j = 0; j < 8; ++j) { float v = c[j]; lds[wave][j + 8 * hi][ln] = (_Float16)(v > 0.f ? v : 0.f); }
  __syncthreads();
  v16h a2 = {};
  if (hi == 0) {
    const v8h t = *(const v8h*)&lds[wave][ln][0];
#pragma unroll
    for (int i = 0; i < 6; ++i) a2[i] = t[i];
    a2[6] = (_Float16)1.f;
  }
  v8f c2 = {};
  c2 = __builtin_amdgcn_wmma_f32_16x16x32_f16(false, a2, false, ((const v16h*)F2b)[lane], (short)0, c2, false, false);
  if (ln < 8) {                                            // cols 6,7 are exact zeros -> pad
#pragma unroll
    for (int j = 0; j < 8; ++j) eN8[(rg0 + j + 8 * hi) * 8 + ln] = (_Float16)c2[j];
  }
}

// e2v segment-sum (/N) + mlp3 : one wave per (b,n) row
__global__ void k_e2v(const _Float16* __restrict__ eN8, const float* __restrict__ W1,
                      const float* __restrict__ b1, const float* __restrict__ W2,
                      const float* __restrict__ b2, _Float16* __restrict__ vN8) {
  const int w = blockIdx.x * 8 + (threadIdx.x >> 5);       // 0..2047
  const int lane = threadIdx.x & 31;
  const int b = w >> 8, n = w & 255;
  const v8h* ep = (const v8h*)eN8 + ((long)b * E_ + (long)n * (N_ - 1));
  float acc[H_] = {0, 0, 0, 0, 0, 0};
  for (int e = lane; e < N_ - 1; e += 32) {
    const v8h t = ep[e];
#pragma unroll
    for (int h = 0; h < H_; ++h) acc[h] += (float)t[h];
  }
#pragma unroll
  for (int off = 16; off >= 1; off >>= 1)
#pragma unroll
    for (int h = 0; h < H_; ++h) acc[h] += __shfl_xor(acc[h], off, 32);
  if (lane == 0) {
    float v1[H_];
#pragma unroll
    for (int c = 0; c < H_; ++c) {
      float s = b1[c];
#pragma unroll
      for (int k = 0; k < H_; ++k) s += (acc[k] * (1.f / (float)N_)) * W1[k * H_ + c];
      v1[c] = s > 0.f ? s : 0.f;
    }
#pragma unroll
    for (int c = 0; c < H_; ++c) {
      float s = b2[c];
#pragma unroll
      for (int k = 0; k < H_; ++k) s += v1[k] * W2[k * H_ + c];
      vN8[w * 8 + c] = (_Float16)s;
    }
    vN8[w * 8 + 6] = (_Float16)0.f; vN8[w * 8 + 7] = (_Float16)0.f;
  }
}

// encoder mlp4 (K = [v_recv|v_send|skip_e|1] = 19) + out linear + gumbel hard
__global__ void k_mlp4(const _Float16* __restrict__ vN8, const _Float16* __restrict__ eN8,
                       const _Float16* __restrict__ F4a, const _Float16* __restrict__ F4b,
                       const float* __restrict__ Wo, const float* __restrict__ bo,
                       const float* __restrict__ gn,
                       float* __restrict__ outEdges, float* __restrict__ outProb,
                       float* __restrict__ edgesW) {
  __shared__ __align__(16) _Float16 lds[4][16][16];
  const int wave = threadIdx.x >> 5, lane = threadIdx.x & 31;
  const int hi = lane >> 4, ln = lane & 15;
  const long rg0 = ((long)blockIdx.x * 4 + wave) * 16;
  const long rg = rg0 + ln;
  const int b = (int)(rg / E_), e = (int)(rg % E_);
  int recv, send; edge_rs(e, recv, send);
  const v8h* vp = (const v8h*)vN8;
  const v8h vr = vp[b * N_ + recv];
  const v8h vs = vp[b * N_ + send];
  const v8h es = ((const v8h*)eN8)[rg];
  v16h a = {};
  if (hi == 0) {                                           // k0-7: vr[0..5], vs[0..1];  k16-23: e[4],e[5],1,0..
#pragma unroll
    for (int i = 0; i < 6; ++i) a[i] = vr[i];
    a[6] = vs[0]; a[7] = vs[1];
    a[8] = es[4]; a[9] = es[5]; a[10] = (_Float16)1.f;
  } else {                                                 // k8-15: vs[2..5], e[0..3];  k24-31: 0
#pragma unroll
    for (int i = 0; i < 4; ++i) a[i] = vs[2 + i];
#pragma unroll
    for (int i = 0; i < 4; ++i) a[4 + i] = es[i];
  }
  v8f c = {};
  c = __builtin_amdgcn_wmma_f32_16x16x32_f16(false, a, false, ((const v16h*)F4a)[lane], (short)0, c, false, false);
#pragma unroll
  for (int j = 0; j < 8; ++j) { float v = c[j]; lds[wave][j + 8 * hi][ln] = (_Float16)(v > 0.f ? v : 0.f); }
  __syncthreads();
  v16h a2 = {};
  if (hi == 0) {
    const v8h t = *(const v8h*)&lds[wave][ln][0];
#pragma unroll
    for (int i = 0; i < 6; ++i) a2[i] = t[i];
    a2[6] = (_Float16)1.f;
  }
  v8f c2 = {};
  c2 = __builtin_amdgcn_wmma_f32_16x16x32_f16(false, a2, false, ((const v16h*)F4b)[lane], (short)0, c2, false, false);
#pragma unroll
  for (int j = 0; j < 8; ++j) lds[wave][j + 8 * hi][ln] = (_Float16)c2[j];   // e2 -> LDS
  __syncthreads();
  if (hi == 0) {
    const int m = ln; const long rr = rg0 + m;
    float l0 = bo[0], l1 = bo[1];
#pragma unroll
    for (int k = 0; k < H_; ++k) {
      const float v = (float)lds[wave][m][k];
      l0 += v * Wo[k * 2 + 0]; l1 += v * Wo[k * 2 + 1];
    }
    const float z0 = (l0 + gn[rr * 2 + 0]) * 2.f;          // 1/tau
    const float z1 = (l1 + gn[rr * 2 + 1]) * 2.f;
    const float h0 = (z0 >= z1) ? 1.f : 0.f;               // straight-through hard value
    const float pm = fmaxf(l0, l1);
    const float p0 = __expf(l0 - pm), p1 = __expf(l1 - pm);
    const float inv = 1.f / (p0 + p1);
    outEdges[rr * 2 + 0] = h0;       outEdges[rr * 2 + 1] = 1.f - h0;
    outProb [rr * 2 + 0] = p0 * inv; outProb [rr * 2 + 1] = p1 * inv;
    edgesW  [rr * 2 + 0] = h0;       edgesW  [rr * 2 + 1] = 1.f - h0;
  }
}

// decoder per-step messages: both edge types in one WMMA pair; f16 padded output rows
__global__ void k_msgs(const _Float16* __restrict__ hidh8, const float* __restrict__ edg,
                       const _Float16* __restrict__ FMa, const _Float16* __restrict__ FMb,
                       _Float16* __restrict__ msgs8) {
  __shared__ __align__(16) _Float16 lds[4][16][16];
  const int wave = threadIdx.x >> 5, lane = threadIdx.x & 31;
  const int hi = lane >> 4, ln = lane & 15;
  const long rg0 = ((long)blockIdx.x * 4 + wave) * 16;
  const long rg = rg0 + ln;
  const int b = (int)(rg / E_), e = (int)(rg % E_);
  int recv, send; edge_rs(e, recv, send);
  const v8h* hp = (const v8h*)hidh8;
  const v8h hs = hp[b * N_ + send];                        // decoder order: [send|recv]
  const v8h hr = hp[b * N_ + recv];
  v16h a = {};
  if (hi == 0) {
#pragma unroll
    for (int i = 0; i < 6; ++i) a[i] = hs[i];
    a[6] = hr[0]; a[7] = hr[1];
  } else {
#pragma unroll
    for (int i = 0; i < 4; ++i) a[i] = hr[2 + i];
    a[4] = (_Float16)1.f;
  }
  v8f c = {};
  c = __builtin_amdgcn_wmma_f32_16x16x32_f16(false, a, false, ((const v16h*)FMa)[lane], (short)0, c, false, false);
#pragma unroll
  for (int j = 0; j < 8; ++j) lds[wave][j + 8 * hi][ln] = (_Float16)fast_tanh(c[j]);
  __syncthreads();
  const v8h t = *(const v8h*)&lds[wave][ln][hi * 8];       // one ds_load_b128
  v16h a2 = {};
#pragma unroll
  for (int i = 0; i < 8; ++i) a2[i] = t[i];
  if (hi) a2[4] = (_Float16)1.f;                           // k=12 bias row (lds there is 0)
  v8f c2 = {};
  c2 = __builtin_amdgcn_wmma_f32_16x16x32_f16(false, a2, false, ((const v16h*)FMb)[lane], (short)0, c2, false, false);
#pragma unroll
  for (int j = 0; j < 8; ++j) {
    const float tv = fast_tanh(c2[j]);
    const int src = hi * 16 + ((ln < 6) ? ln + 6 : ln - 6);
    const float other = __shfl(tv, src, 32);               // pair type0/type1
    float mv = 0.f;
    if (ln < 6) {
      const float2 ev = ((const float2*)edg)[rg0 + j + 8 * hi];
      mv = 0.5f * (tv * ev.x + other * ev.y);              // /ET, hard one-hot weights
    }
    if (ln < 8) lds[wave][j + 8 * hi][ln] = (_Float16)mv;  // transpose tile (cols 6,7 = 0)
  }
  __syncthreads();
  if (hi == 0) {                                           // one b128 store per row
    const v8h row = *(const v8h*)&lds[wave][ln][0];
    ((v8h*)msgs8)[rg0 + ln] = row;
  }
}

// decoder per-step node update: wave-per-row vector reduce + GRU + output MLP
__global__ void k_node(const float* __restrict__ x, const _Float16* __restrict__ msgs8,
                       float* __restrict__ hid, _Float16* __restrict__ hidh8,
                       const float* __restrict__ Wr, const float* __restrict__ br,
                       const float* __restrict__ Wi, const float* __restrict__ bi,
                       const float* __restrict__ Wn, const float* __restrict__ bn,
                       const float* __restrict__ Hr, const float* __restrict__ Hi,
                       const float* __restrict__ Hh,
                       const float* __restrict__ O1, const float* __restrict__ o1b,
                       const float* __restrict__ O2, const float* __restrict__ o2b,
                       const float* __restrict__ O3, const float* __restrict__ o3b,
                       float* __restrict__ preds, int t) {
  const int w = blockIdx.x * 8 + (threadIdx.x >> 5);       // 0..2047
  const int lane = threadIdx.x & 31;
  const int b = w >> 8, n = w & 255;
  const v8h* mp = (const v8h*)msgs8 + ((long)b * E_ + (long)n * (N_ - 1));
  float acc[H_] = {0, 0, 0, 0, 0, 0};
  for (int e = lane; e < N_ - 1; e += 32) {
    const v8h u = mp[e];
#pragma unroll
    for (int h = 0; h < H_; ++h) acc[h] += (float)u[h];
  }
#pragma unroll
  for (int off = 16; off >= 1; off >>= 1)
#pragma unroll
    for (int h = 0; h < H_; ++h) acc[h] += __shfl_xor(acc[h], off, 32);
  if (lane == 0) {
    float agg[H_];
#pragma unroll
    for (int h = 0; h < H_; ++h) agg[h] = acc[h] * (1.f / (float)D_);
    const float4* xp = (const float4*)(x + ((long)(b * T_ + t) * N_ + n) * D_);
    const float4 x0 = xp[0], x1 = xp[1], x2 = xp[2];
    float ins[D_] = {x0.x, x0.y, x0.z, x0.w, x1.x, x1.y, x1.z, x1.w, x2.x, x2.y, x2.z, x2.w};
    float hnew[H_];
#pragma unroll
    for (int c = 0; c < H_; ++c) {
      float sr = br[c], si = bi[c], sn = bn[c];
#pragma unroll
      for (int d = 0; d < D_; ++d) {
        sr += ins[d] * Wr[d * H_ + c]; si += ins[d] * Wi[d * H_ + c]; sn += ins[d] * Wn[d * H_ + c];
      }
      float ar = 0.f, ai = 0.f, ah = 0.f;
#pragma unroll
      for (int k = 0; k < H_; ++k) {
        ar += agg[k] * Hr[k * H_ + c]; ai += agg[k] * Hi[k * H_ + c]; ah += agg[k] * Hh[k * H_ + c];
      }
      const float rr = sigmoidf_(sr + ar);
      const float ii = sigmoidf_(si + ai);
      const float nn = fast_tanh(sn + rr * ah);
      hnew[c] = (1.f - ii) * nn + ii * hid[w * H_ + c];
    }
    float t1[H_], t2[H_], p[D_];
#pragma unroll
    for (int c = 0; c < H_; ++c) {
      float s = o1b[c];
#pragma unroll
      for (int k = 0; k < H_; ++k) s += hnew[k] * O1[k * H_ + c];
      t1[c] = s > 0.f ? s : 0.f;
    }
#pragma unroll
    for (int c = 0; c < H_; ++c) {
      float s = o2b[c];
#pragma unroll
      for (int k = 0; k < H_; ++k) s += t1[k] * O2[k * H_ + c];
      t2[c] = s > 0.f ? s : 0.f;
    }
#pragma unroll
    for (int d = 0; d < D_; ++d) {
      float s = o3b[d];
#pragma unroll
      for (int k = 0; k < H_; ++k) s += t2[k] * O3[k * D_ + d];
      p[d] = ins[d] + s;
    }
    float4* pp = (float4*)(preds + ((long)(b * (T_ - 1) + t) * N_ + n) * D_);
    pp[0] = make_float4(p[0], p[1], p[2], p[3]);
    pp[1] = make_float4(p[4], p[5], p[6], p[7]);
    pp[2] = make_float4(p[8], p[9], p[10], p[11]);
#pragma unroll
    for (int c = 0; c < H_; ++c) { hid[w * H_ + c] = hnew[c]; hidh8[w * 8 + c] = (_Float16)hnew[c]; }
    hidh8[w * 8 + 6] = (_Float16)0.f; hidh8[w * 8 + 7] = (_Float16)0.f;
  }
}

// ---------------- host launcher ----------------
extern "C" void kernel_launch(void* const* d_in, const int* in_sizes, int n_in,
                              void* d_out, int out_size, void* d_ws, size_t ws_size,
                              hipStream_t stream) {
  const float* x   = (const float*)d_in[0];
  // d_in[1]/d_in[2]: 128MB of one-hot rel_rec/rel_send -- derived analytically, never read.
  const float* gn  = (const float*)d_in[3];
  const float* m1W1 = (const float*)d_in[4];  const float* m1b1 = (const float*)d_in[5];
  const float* m1W2 = (const float*)d_in[6];  const float* m1b2 = (const float*)d_in[7];
  const float* m2W1 = (const float*)d_in[8];  const float* m2b1 = (const float*)d_in[9];
  const float* m2W2 = (const float*)d_in[10]; const float* m2b2 = (const float*)d_in[11];
  const float* m3W1 = (const float*)d_in[12]; const float* m3b1 = (const float*)d_in[13];
  const float* m3W2 = (const float*)d_in[14]; const float* m3b2 = (const float*)d_in[15];
  const float* m4W1 = (const float*)d_in[16]; const float* m4b1 = (const float*)d_in[17];
  const float* m4W2 = (const float*)d_in[18]; const float* m4b2 = (const float*)d_in[19];
  const float* Wo   = (const float*)d_in[20]; const float* bo   = (const float*)d_in[21];
  const float* W1m = (const float*)d_in[22];  const float* b1m = (const float*)d_in[23];
  const float* W2m = (const float*)d_in[24];  const float* b2m = (const float*)d_in[25];
  const float* Hr  = (const float*)d_in[26];  const float* Hi  = (const float*)d_in[27];
  const float* Hh  = (const float*)d_in[28];
  const float* Wr  = (const float*)d_in[29];  const float* br  = (const float*)d_in[30];
  const float* Wi  = (const float*)d_in[31];  const float* bi  = (const float*)d_in[32];
  const float* Wn  = (const float*)d_in[33];  const float* bn  = (const float*)d_in[34];
  const float* O1  = (const float*)d_in[35];  const float* o1b = (const float*)d_in[36];
  const float* O2  = (const float*)d_in[37];  const float* o2b = (const float*)d_in[38];
  const float* O3  = (const float*)d_in[39];  const float* o3b = (const float*)d_in[40];

  float* outPreds = (float*)d_out;
  float* outEdges = outPreds + PRED_SZ;
  float* outProb  = outEdges + EDGE_SZ;

  char* ws = (char*)d_ws;
  auto align256 = [](size_t v) { return (v + 255) & ~(size_t)255; };
  size_t off = 0;
  _Float16* xe16  = (_Float16*)(ws + off); off = align256(off + (size_t)B_ * N_ * XE_K * 2);
  _Float16* hN8   = (_Float16*)(ws + off); off = align256(off + (size_t)B_ * N_ * 8 * 2);
  _Float16* eN8   = (_Float16*)(ws + off); off = align256(off + (size_t)ROWS * 8 * 2);
  _Float16* vN8   = (_Float16*)(ws + off); off = align256(off + (size_t)B_ * N_ * 8 * 2);
  float*    edgW  = (float*)   (ws + off); off = align256(off + (size_t)ROWS * ET_ * 4);
  float*    hid   = (float*)   (ws + off); off = align256(off + (size_t)B_ * N_ * H_ * 4);
  _Float16* hidh8 = (_Float16*)(ws + off); off = align256(off + (size_t)B_ * N_ * 8 * 2);
  _Float16* msgs8 = (_Float16*)(ws + off); off = align256(off + (size_t)ROWS * 8 * 2);
  _Float16* FB1   = (_Float16*)(ws + off); off = align256(off + (size_t)19 * 32 * 16 * 2);
  _Float16* F2a   = (_Float16*)(ws + off); off = align256(off + (size_t)32 * 16 * 2);
  _Float16* F2b   = (_Float16*)(ws + off); off = align256(off + (size_t)32 * 16 * 2);
  _Float16* F4a   = (_Float16*)(ws + off); off = align256(off + (size_t)32 * 16 * 2);
  _Float16* F4b   = (_Float16*)(ws + off); off = align256(off + (size_t)32 * 16 * 2);
  _Float16* FMa   = (_Float16*)(ws + off); off = align256(off + (size_t)32 * 16 * 2);
  _Float16* FMb   = (_Float16*)(ws + off); off = align256(off + (size_t)32 * 16 * 2);
  (void)ws_size; (void)n_in; (void)in_sizes; (void)out_size;

  const int EDGE_BLOCKS = NTILES / 4;   // 8160, exact

  k_prep<<<4, 256, 0, stream>>>(m1W1, m1b1, m2W1, m2b1, m2W2, m2b2, m4W1, m4b1, m4W2, m4b2,
                                W1m, b1m, W2m, b2m, FB1, F2a, F2b, F4a, F4b, FMa, FMb);
  k_init<<<(B_ * N_ * 8 + 255) / 256, 256, 0, stream>>>(hid, hidh8);
  k_xpose<<<(B_ * N_ * XE_K) / 256, 256, 0, stream>>>(x, xe16);
  k_mlp1<<<(B_ * N_ / 16) / 4, 128, 0, stream>>>(xe16, FB1, m1W2, m1b2, hN8);
  k_mlp2<<<EDGE_BLOCKS, 128, 0, stream>>>(hN8, F2a, F2b, eN8);
  k_e2v<<<B_ * N_ / 8, 256, 0, stream>>>(eN8, m3W1, m3b1, m3W2, m3b2, vN8);
  k_mlp4<<<EDGE_BLOCKS, 128, 0, stream>>>(vN8, eN8, F4a, F4b, Wo, bo, gn,
                                          outEdges, outProb, edgW);
  for (int t = 0; t < T_ - 1; ++t) {
    k_msgs<<<EDGE_BLOCKS, 128, 0, stream>>>(hidh8, edgW, FMa, FMb, msgs8);
    k_node<<<B_ * N_ / 8, 256, 0, stream>>>(x, msgs8, hid, hidh8,
        Wr, br, Wi, bi, Wn, bn, Hr, Hi, Hh, O1, o1b, O2, o2b, O3, o3b, outPreds, t);
  }
}
